// RelationLayer_64871186038957
// MI455X (gfx1250) — matrix operand order repaired and checked by reading, not compiled, not run on hardware
//
#include <hip/hip_runtime.h>

typedef __attribute__((ext_vector_type(16))) _Float16 v16h;
typedef __attribute__((ext_vector_type(8)))  float    v8f;
typedef __attribute__((ext_vector_type(4)))  unsigned int v4u;
typedef __attribute__((ext_vector_type(8)))  int      v8i_;
typedef __attribute__((ext_vector_type(4)))  int      v4i_;

#define FLOAT_EPS 1.1920928955078125e-07f

#define B_  16
#define P_  8192
#define I_  256
#define O_  128
#define L_  256
#define IO_ 32768   // I_*O_

#if defined(__has_builtin)
#if __has_builtin(__builtin_amdgcn_tensor_load_to_lds) && \
    __has_builtin(__builtin_amdgcn_s_wait_tensorcnt)
#define HAVE_TDM 1
#endif
#endif

__device__ __forceinline__ float wave_max(float v) {
  #pragma unroll
  for (int off = 16; off >= 1; off >>= 1) v = fmaxf(v, __shfl_xor(v, off, 32));
  return v;
}
__device__ __forceinline__ float wave_sum(float v) {
  #pragma unroll
  for (int off = 16; off >= 1; off >>= 1) v += __shfl_xor(v, off, 32);
  return v;
}
__device__ __forceinline__ float clip01(float v) { return fminf(fmaxf(v, 0.0f), 1.0f); }

// ---------------------------------------------------------------------------
// Kernel 1: logits[k][b][i*128+o] = sum_l latent[b][l] * comp_k[l][i][o]
// GEMM with M = 16 (batch), K = 256 (l), N = 32768 (i*o), via WMMA f16.
// comp is streamed exactly once -> non-temporal loads (keep L2 clean).
// grid = (256, 2), block = 256 threads (8 waves); wave w owns N-tile of 16.
// ---------------------------------------------------------------------------
__global__ __launch_bounds__(256) void k_logits(
    const float* __restrict__ latent,
    const float* __restrict__ comp1,
    const float* __restrict__ comp2,
    float* __restrict__ logits)
{
  const int kcomp = blockIdx.y;
  const float* __restrict__ comp = kcomp ? comp2 : comp1;

  __shared__ _Float16 Ls[16 * 258];           // 16 x 256 latent, padded rows

  const int tid = threadIdx.x;
  {
    const int row = tid >> 4;
    const int c0  = (tid & 15) * 16;
    const float* src = latent + row * L_ + c0;
    #pragma unroll
    for (int j = 0; j < 16; ++j) Ls[row * 258 + c0 + j] = (_Float16)src[j];
  }
  __syncthreads();

  const int wave = tid >> 5;
  const int lane = tid & 31;
  const int n    = blockIdx.x * 128 + wave * 16 + (lane & 15);  // global N col
  const int m    = lane & 15;              // A-matrix row (batch index group)
  const int kbA  = (lane >> 4) * 16;       // A-frag K base (contiguous K, matches B)
  const int kbB  = (lane >> 4) * 16;       // B-frag K base

  v8f acc = {};
  #pragma unroll
  for (int kc = 0; kc < L_; kc += 32) {
    v16h a, b;
    #pragma unroll
    for (int t = 0; t < 4; ++t) {
      a[2*t]     = Ls[m * 258 + kc + kbA + 2*t];
      a[2*t + 1] = Ls[m * 258 + kc + kbA + 2*t + 1];
      a[8 + 2*t]     = Ls[m * 258 + kc + kbA + 8 + 2*t];
      a[8 + 2*t + 1] = Ls[m * 258 + kc + kbA + 8 + 2*t + 1];
    }
    #pragma unroll
    for (int t = 0; t < 8; ++t) {
      b[2*t]     = (_Float16)__builtin_nontemporal_load(
                       comp + (size_t)(kc + kbB + 2*t)     * IO_ + n);
      b[2*t + 1] = (_Float16)__builtin_nontemporal_load(
                       comp + (size_t)(kc + kbB + 2*t + 1) * IO_ + n);
    }
    acc = __builtin_amdgcn_wmma_f32_16x16x32_f16(
        false, a, false, b, (short)0, acc, false, false);
  }

  // C layout: VGPR r, lane -> M = r + 8*(lane>>4), N = lane&15 (folded into n)
  const int mb = 8 * (lane >> 4);
  #pragma unroll
  for (int r = 0; r < 8; ++r) {
    const int bidx = r + mb;                // batch index = matrix row
    logits[(size_t)(kcomp * 16 + bidx) * IO_ + n] = acc[r];
  }
}

// ---------------------------------------------------------------------------
// Kernel 2: softmax over i, Gumbel perturbation, second softmax over i.
// One wave per (k, b, o) column; 256 i's -> 8 per lane + shuffle reductions.
// Emits mask_s as f16 packed [b][i][k*128+o]  (the phase-3 B-matrix).
// grid = (16, 16, 2), block = 256 (8 waves -> 8 o's per block).
// ---------------------------------------------------------------------------
__global__ __launch_bounds__(256) void k_gumbel(
    const float* __restrict__ logits,
    const float* __restrict__ u,
    const float* __restrict__ temperature,
    _Float16* __restrict__ Wm)
{
  const int o     = blockIdx.x * 8 + (threadIdx.x >> 5);
  const int b     = blockIdx.y;
  const int kcomp = blockIdx.z;
  const int lane  = threadIdx.x & 31;

  float t = temperature[0];
  t = fminf(fmaxf(t, FLOAT_EPS), 2.0f);
  t = fmaxf(t, 0.001f);
  const float inv_t = 1.0f / t;

  const float* zcol = logits + (size_t)(kcomp * 16 + b) * IO_ + o;

  float z[8];
  #pragma unroll
  for (int ii = 0; ii < 8; ++ii) z[ii] = zcol[(size_t)(lane + 32 * ii) * O_];

  float m1 = z[0];
  #pragma unroll
  for (int ii = 1; ii < 8; ++ii) m1 = fmaxf(m1, z[ii]);
  m1 = wave_max(m1);

  float e[8], s1 = 0.0f;
  #pragma unroll
  for (int ii = 0; ii < 8; ++ii) { e[ii] = __expf(z[ii] - m1); s1 += e[ii]; }
  s1 = wave_sum(s1);
  const float inv_s1 = 1.0f / s1;

  const float* ucol = u + (size_t)(b * 2 + kcomp) * (I_ * O_) + o;

  float y[8];
  #pragma unroll
  for (int ii = 0; ii < 8; ++ii) {
    const int i = lane + 32 * ii;
    const float mask = e[ii] * inv_s1;
    const float uv   = fmaxf(ucol[(size_t)i * O_], FLOAT_EPS);
    const float g    = -__logf(-__logf(uv));
    y[ii] = (__logf(mask + FLOAT_EPS) + g) * inv_t;
  }

  float m2 = y[0];
  #pragma unroll
  for (int ii = 1; ii < 8; ++ii) m2 = fmaxf(m2, y[ii]);
  m2 = wave_max(m2);

  float w[8], s2 = 0.0f;
  #pragma unroll
  for (int ii = 0; ii < 8; ++ii) { w[ii] = __expf(y[ii] - m2); s2 += w[ii]; }
  s2 = wave_sum(s2);
  const float inv_s2 = 1.0f / s2;

  #pragma unroll
  for (int ii = 0; ii < 8; ++ii) {
    const int i = lane + 32 * ii;
    Wm[(size_t)(b * I_ + i) * 256 + kcomp * O_ + o] = (_Float16)(w[ii] * inv_s2);
  }
}

// ---------------------------------------------------------------------------
// Kernel 3: per batch b: C(8192x256) = X_b(8192x256 f32->f16) * W_b(256x256 f16)
// fused epilogue: a = C[:, :128], bb = C[:, 128:],
// out = clip([a+bb, a+bb-1, a-bb, bb-a]), non-temporal stores.
// W chunk staged in LDS by the Tensor Data Mover (D# per ISA ch.8):
//   2D tile 256 halves x 32 rows, TDM pad_enable adds 4 DWORDs per 128 DWORDs
//   -> LDS row stride 264 halves (bank-conflict padding done by the DMA).
// Issued by wave 0 only (TDM is wave-level, EXEC ignored) and overlapped with
// the X-tile f32->f16 staging; completion via s_wait_tensorcnt + barrier.
// grid = (64, 16), block = 256.
// ---------------------------------------------------------------------------
__global__ __launch_bounds__(256) void k_sel_out(
    const float* __restrict__ x,
    const _Float16* __restrict__ Wm,
    float* __restrict__ out)
{
  const int bidx   = blockIdx.y;
  const int p_base = blockIdx.x * 128;

  __shared__ _Float16 Xs[128 * 34];   // 128 rows x 32 K, padded (17-bank stride)
  __shared__ _Float16 Ws[32 * 264];   // 32 K-rows x 256 N, padded (TDM pad)

  const int tid  = threadIdx.x;
  const int wave = tid >> 5;
  const int lane = tid & 31;
  const int m    = lane & 15;
  const int kbA  = (lane >> 4) * 16;
  const int kbB  = (lane >> 4) * 16;
  const int nA   = wave * 16 + (lane & 15);   // a-column (0..127); b-col = nA+128

  v8f accA[8] = {};
  v8f accB[8] = {};

  const float* __restrict__ xbase =
      x + (size_t)(bidx * P_ + p_base) * (size_t)I_;
  const _Float16* __restrict__ WmB = Wm + (size_t)bidx * I_ * 256;

  for (int kc = 0; kc < I_; kc += 32) {
    __syncthreads();

#ifdef HAVE_TDM
    if (wave == 0) {
      // ---- Tensor DMA descriptor (D#) -------------------------------------
      const unsigned long long gaddr =
          (unsigned long long)(uintptr_t)(WmB + (size_t)kc * 256);
      const unsigned int laddr = (unsigned int)(uintptr_t)&Ws[0];
      v4u g0;
      g0[0] = 1u;                                        // count=1, user D#
      g0[1] = laddr;                                     // lds_addr
      g0[2] = (unsigned int)(gaddr & 0xFFFFFFFFu);       // global_addr[31:0]
      g0[3] = (unsigned int)((gaddr >> 32) & 0x01FFFFFFu)
            | (2u << 30);                                // addr[56:32], type=2
      v8i_ g1;
      g1[0] = (1 << 16)        // data_size = 2 bytes
            | (1 << 20)        // pad_enable
            | (6 << 22)        // pad_interval: every 128 DWORDs (one 256h row)
            | (3 << 25);       // pad_amount: 4 DWORDs (8 halves) -> stride 264
      g1[1] = (256 & 0xFFFF) << 16;          // tensor_dim0[15:0] = 256
      g1[2] = (256 << 16);                   // tensor_dim0 hi=0 | tensor_dim1=256
      g1[3] = (256 << 16);                   // tensor_dim1 hi=0 | tile_dim0=256
      g1[4] = 32;                            // tile_dim1=32, tile_dim2=0
      g1[5] = 256;                           // tensor_dim0_stride[31:0] = 256
      g1[6] = 0;                             // stride0 hi | tensor_dim1_stride lo
      g1[7] = 0;
      v4i_ gz = {};
#if defined(__clang_major__) && (__clang_major__ >= 23)
      v8i_ gz8 = {};
      __builtin_amdgcn_tensor_load_to_lds(g0, g1, gz, gz, gz8, 0);
#else
      __builtin_amdgcn_tensor_load_to_lds(g0, g1, gz, gz, 0);
#endif
    }
#endif

    {   // stage X chunk: 128 rows x 32 halves (NT f32 loads, cvt -> f16)
      const int row = tid >> 1;
      const int c0  = (tid & 1) * 16;
      const float* src = xbase + (size_t)row * I_ + kc + c0;
      #pragma unroll
      for (int j = 0; j < 16; ++j)
        Xs[row * 34 + c0 + j] = (_Float16)__builtin_nontemporal_load(src + j);
    }

#ifdef HAVE_TDM
    if (wave == 0) __builtin_amdgcn_s_wait_tensorcnt(0);
#else
    {   // fallback: stage W chunk manually (rows kc..kc+31, 256 halves each)
      const int kr  = tid >> 3;
      const int seg = tid & 7;
      const unsigned int* src =
          (const unsigned int*)(WmB + (size_t)(kc + kr) * 256) + seg * 16;
      unsigned int* dst = (unsigned int*)&Ws[kr * 264] + seg * 16;
      #pragma unroll
      for (int j = 0; j < 16; ++j) dst[j] = src[j];
    }
#endif
    __syncthreads();

    v16h bA, bB;
    #pragma unroll
    for (int t = 0; t < 8; ++t) {
      bA[2*t]     = Ws[(kbB + 2*t)     * 264 + nA];
      bA[2*t + 1] = Ws[(kbB + 2*t + 1) * 264 + nA];
      bB[2*t]     = Ws[(kbB + 2*t)     * 264 + 128 + nA];
      bB[2*t + 1] = Ws[(kbB + 2*t + 1) * 264 + 128 + nA];
    }

    #pragma unroll
    for (int ms = 0; ms < 8; ++ms) {
      const int rowm = ms * 16 + m;
      v16h a;
      #pragma unroll
      for (int t = 0; t < 4; ++t) {
        a[2*t]     = Xs[rowm * 34 + kbA + 2*t];
        a[2*t + 1] = Xs[rowm * 34 + kbA + 2*t + 1];
        a[8 + 2*t]     = Xs[rowm * 34 + kbA + 8 + 2*t];
        a[8 + 2*t + 1] = Xs[rowm * 34 + kbA + 8 + 2*t + 1];
      }
      accA[ms] = __builtin_amdgcn_wmma_f32_16x16x32_f16(
          false, a, false, bA, (short)0, accA[ms], false, false);
      accB[ms] = __builtin_amdgcn_wmma_f32_16x16x32_f16(
          false, a, false, bB, (short)0, accB[ms], false, false);
    }
  }

  // fused epilogue: C layout -> row = r + 8*(lane>>4), col = lane&15
  const int mb = 8 * (lane >> 4);
  const int o  = wave * 16 + (lane & 15);
  #pragma unroll
  for (int ms = 0; ms < 8; ++ms) {
    #pragma unroll
    for (int r = 0; r < 8; ++r) {
      const int p = p_base + ms * 16 + r + mb;
      const float av = accA[ms][r];
      const float bv = accB[ms][r];
      float* orow = out + (size_t)(bidx * P_ + p) * 512;
      __builtin_nontemporal_store(clip01(av + bv),        orow + o);
      __builtin_nontemporal_store(clip01(av + bv - 1.0f), orow + o + 128);
      __builtin_nontemporal_store(clip01(av - bv),        orow + o + 256);
      __builtin_nontemporal_store(clip01(bv - av),        orow + o + 384);
    }
  }
}

// ---------------------------------------------------------------------------
extern "C" void kernel_launch(void* const* d_in, const int* in_sizes, int n_in,
                              void* d_out, int out_size, void* d_ws, size_t ws_size,
                              hipStream_t stream) {
  const float* x      = (const float*)d_in[0];
  const float* latent = (const float*)d_in[1];
  const float* comp1  = (const float*)d_in[2];
  const float* comp2  = (const float*)d_in[3];
  const float* temp   = (const float*)d_in[4];
  const float* u      = (const float*)d_in[5];
  float* out = (float*)d_out;

  // workspace: logits (2*16*32768 f32 = 4 MB) then packed mask_s f16 (2 MB)
  float*    logits = (float*)d_ws;
  _Float16* Wm     = (_Float16*)((char*)d_ws + (size_t)2 * B_ * IO_ * sizeof(float));

  k_logits<<<dim3(256, 2), 256, 0, stream>>>(latent, comp1, comp2, logits);
  k_gumbel<<<dim3(16, 16, 2), 256, 0, stream>>>(logits, u, temp, Wm);
  k_sel_out<<<dim3(64, 16), 256, 0, stream>>>(x, Wm, out);
}
